// MultiHeadedSelfAttentionModule_47596827574545
// MI455X (gfx1250) — matrix operand order repaired and checked by reading, not compiled, not run on hardware
//
#include <hip/hip_runtime.h>
#include <hip/hip_bf16.h>

// ---------------------------------------------------------------------------
// Conformer relative-position MHSA for MI455X (gfx1250, wave32, WMMA).
// All GEMMs use v_wmma_f32_16x16x32_bf16 (bf16 in, f32 accumulate).
// Workspace requirement: ~66 MB.
// ---------------------------------------------------------------------------

#define BATCH 8
#define SEQ   1024
#define DMODEL 512
#define NHEAD 8
#define DHEAD 64
#define INV_SQRT_D 0.044194173824159216f  // 1/sqrt(512)

typedef __attribute__((ext_vector_type(16))) __bf16 v16bf;
typedef __attribute__((ext_vector_type(8)))  float  v8f;

// ---------------------------------------------------------------------------
// WMMA fragment helpers (wave32 layouts per CDNA5 ISA §7.12.2)
// ---------------------------------------------------------------------------

// A fragment: 16x32 bf16, rows m0..m0+15, K columns k0..k0+31, row-major src.
// Lane L: row = L%16.  half = L/16.
// v16bf elements [0..7]  = K (8*half .. 8*half+7)        (VGPRs 0..3)
//                [8..15] = K (16+8*half .. 16+8*half+7)  (VGPRs 4..7)
__device__ inline v16bf load_a_frag(const __bf16* __restrict__ A, int lda,
                                    int m0, int k0, int lane) {
  const int row  = m0 + (lane & 15);
  const int half = lane >> 4;
  const __bf16* p0 = A + (size_t)row * lda + k0 + 8 * half;
  const __bf16* p1 = p0 + 16;
  v16bf a;
#pragma unroll
  for (int i = 0; i < 8; ++i) { a[i] = p0[i]; a[i + 8] = p1[i]; }
  return a;
}

// B fragment: 32x16 bf16 where B[k][n] = BT[n][k], BT row-major [N][K].
// Lane L: col = L%16, half = L/16; elements [0..7] = K (8*half + i), [8..15] = K (16 + 8*half + i).
__device__ inline v16bf load_bt_frag(const __bf16* __restrict__ BT, int ldb,
                                     int k0, int n0, int lane) {
  const int col  = n0 + (lane & 15);
  const int half = lane >> 4;
  const __bf16* p0 = BT + (size_t)col * ldb + k0 + 8 * half;
  const __bf16* p1 = p0 + 16;
  v16bf b;
#pragma unroll
  for (int i = 0; i < 8; ++i) { b[i] = p0[i]; b[i + 8] = p1[i]; }
  return b;
}

__device__ inline v8f wmma_bf16(v16bf a, v16bf b, v8f c) {
  // (neg_a, A, neg_b, B, c_mod, C, reuse_a, reuse_b)
  return __builtin_amdgcn_wmma_f32_16x16x32_bf16(
      false, a, false, b, (short)0, c, false, false);
}

// Shared inner loop: one wave computes a 16x32 C strip (two 16x16 frags).
__device__ inline void gemm_bf16_tile(const __bf16* __restrict__ A, int lda,
                                      const __bf16* __restrict__ BT, int ldb,
                                      int K, int m0w, int n0w, int lane,
                                      v8f& acc0, v8f& acc1) {
  for (int kk = 0; kk < K; kk += 32) {
    v16bf a  = load_a_frag(A, lda, m0w, kk, lane);
    v16bf b0 = load_bt_frag(BT, ldb, kk, n0w, lane);
    v16bf b1 = load_bt_frag(BT, ldb, kk, n0w + 16, lane);
    acc0 = wmma_bf16(a, b0, acc0);
    acc1 = wmma_bf16(a, b1, acc1);
  }
}

// C/D layout: lane L -> col N = L%16 (lanes 16-31 mirror), VGPR r -> row r + 8*(L/16).

// ---------------------------------------------------------------------------
// Preprocessing kernels
// ---------------------------------------------------------------------------

// WT[n*512+k] = (bf16) W[k*512+n]   (512x512)
__global__ __launch_bounds__(256) void convert_transpose512_kernel(
    const float* __restrict__ W, __bf16* __restrict__ WT) {
  const int idx = blockIdx.x * 256 + threadIdx.x;
  const int n = idx >> 9, k = idx & 511;
  WT[idx] = (__bf16)W[k * 512 + n];
}

__global__ __launch_bounds__(256) void convert_bf16_kernel(
    const float* __restrict__ X, __bf16* __restrict__ Y, int n) {
  const int idx = blockIdx.x * 256 + threadIdx.x;
  if (idx < n) Y[idx] = (__bf16)X[idx];
}

// LayerNorm over D=512 per row; biased variance, eps=1e-5. Output bf16.
__global__ __launch_bounds__(256) void layernorm_bf16_kernel(
    const float* __restrict__ x, const float* __restrict__ gamma,
    const float* __restrict__ beta, __bf16* __restrict__ out) {
  __shared__ float red[256];
  const int row = blockIdx.x;
  const int tid = threadIdx.x;
  const float* xr = x + (size_t)row * DMODEL;
  const float a = xr[tid], b = xr[tid + 256];
  red[tid] = a + b;
  __syncthreads();
  for (int off = 128; off > 0; off >>= 1) {
    if (tid < off) red[tid] += red[tid + off];
    __syncthreads();
  }
  const float mean = red[0] * (1.0f / DMODEL);
  __syncthreads();
  const float da = a - mean, db = b - mean;
  red[tid] = da * da + db * db;
  __syncthreads();
  for (int off = 128; off > 0; off >>= 1) {
    if (tid < off) red[tid] += red[tid + off];
    __syncthreads();
  }
  const float inv = rsqrtf(red[0] * (1.0f / DMODEL) + 1e-5f);
  __bf16* orow = out + (size_t)row * DMODEL;
  orow[tid]       = (__bf16)(da * inv * gamma[tid] + beta[tid]);
  orow[tid + 256] = (__bf16)(db * inv * gamma[tid + 256] + beta[tid + 256]);
}

// ---------------------------------------------------------------------------
// Projection GEMM: C[M x 512] = A_bf16[M x 512] @ WT_bf16[512 x 512]^T (+bias)
// Block = 256 threads = 8 waves; tile 64x64; wave (w&3) -> 16-row sub,
// (w>>2) -> 32-col sub.  Epilogue modes:
//   0: bf16 out0 in per-head layout [B][H][S][DH]        (K proj, P proj)
//   1: bf16 out0/out1 per-head layout, + head biases     (Q proj -> q+u, q+v)
//   2: bf16 out0 transposed per-head [B][H][DH][S]       (V proj)
//   3: f32 outf row-major [M][512]                       (final O proj)
// ---------------------------------------------------------------------------
__global__ __launch_bounds__(256) void wmma_proj_gemm_kernel(
    const __bf16* __restrict__ A, const __bf16* __restrict__ WT,
    const float* __restrict__ bias, const float* __restrict__ hb0,
    const float* __restrict__ hb1, __bf16* __restrict__ out0,
    __bf16* __restrict__ out1, float* __restrict__ outf, int mode) {
  const int lane = threadIdx.x & 31;
  const int wave = threadIdx.x >> 5;
  const int m0w = blockIdx.x * 64 + 16 * (wave & 3);
  const int n0w = blockIdx.y * 64 + 32 * (wave >> 2);

  v8f acc0, acc1;
#pragma unroll
  for (int i = 0; i < 8; ++i) { acc0[i] = 0.0f; acc1[i] = 0.0f; }
  gemm_bf16_tile(A, DMODEL, WT, DMODEL, DMODEL, m0w, n0w, lane, acc0, acc1);

  const int half = lane >> 4;
  const int colb = lane & 15;
#pragma unroll
  for (int r = 0; r < 8; ++r) {
    const int m = m0w + r + 8 * half;
    const int b = m >> 10;        // SEQ = 1024
    const int s = m & (SEQ - 1);
#pragma unroll
    for (int f = 0; f < 2; ++f) {
      const int n = n0w + colb + 16 * f;
      const float acc = (f == 0) ? acc0[r] : acc1[r];
      const float bb = bias ? bias[n] : 0.0f;
      const int h = n >> 6, dh = n & (DHEAD - 1);
      if (mode == 1) {
        const size_t idx = ((((size_t)b * NHEAD + h) * SEQ) + s) * DHEAD + dh;
        out0[idx] = (__bf16)(acc + bb + hb0[n]);
        out1[idx] = (__bf16)(acc + bb + hb1[n]);
      } else if (mode == 0) {
        const size_t idx = ((((size_t)b * NHEAD + h) * SEQ) + s) * DHEAD + dh;
        out0[idx] = (__bf16)(acc + bb);
      } else if (mode == 2) {
        const size_t idx = ((((size_t)b * NHEAD + h) * DHEAD) + dh) * SEQ + s;
        out0[idx] = (__bf16)(acc + bb);
      } else {
        outf[(size_t)m * DMODEL + n] = acc + bb;
      }
    }
  }
}

// ---------------------------------------------------------------------------
// Raw positional scores for one (b,h): R[q][j] = sum_d qv[q][d] * p[j][d]
// M = N = 1024, K = 64.
// ---------------------------------------------------------------------------
__global__ __launch_bounds__(256) void wmma_pos_raw_kernel(
    const __bf16* __restrict__ qv, const __bf16* __restrict__ p,
    float* __restrict__ R) {
  const int lane = threadIdx.x & 31;
  const int wave = threadIdx.x >> 5;
  const int m0w = blockIdx.x * 64 + 16 * (wave & 3);
  const int n0w = blockIdx.y * 64 + 32 * (wave >> 2);
  v8f acc0, acc1;
#pragma unroll
  for (int i = 0; i < 8; ++i) { acc0[i] = 0.0f; acc1[i] = 0.0f; }
  gemm_bf16_tile(qv, DHEAD, p, DHEAD, DHEAD, m0w, n0w, lane, acc0, acc1);

  const int half = lane >> 4;
  const int colb = lane & 15;
#pragma unroll
  for (int r = 0; r < 8; ++r) {
    const int q = m0w + r + 8 * half;
    R[(size_t)q * SEQ + n0w + colb]      = acc0[r];
    R[(size_t)q * SEQ + n0w + colb + 16] = acc1[r];
  }
}

// ---------------------------------------------------------------------------
// Content scores + exact Transformer-XL rel-shift of R, scaled by 1/sqrt(D).
// shifted[q][k] = R[q, k-q+S-1]        if k <= q
//              = 0                     if k == q+1
//              = R[q+1, k-q-2]         if k >  q+1   (exact pad/reshape alias)
// ---------------------------------------------------------------------------
__global__ __launch_bounds__(256) void wmma_score_kernel(
    const __bf16* __restrict__ qu, const __bf16* __restrict__ kmat,
    const float* __restrict__ R, float* __restrict__ Sc) {
  const int lane = threadIdx.x & 31;
  const int wave = threadIdx.x >> 5;
  const int m0w = blockIdx.x * 64 + 16 * (wave & 3);
  const int n0w = blockIdx.y * 64 + 32 * (wave >> 2);
  v8f acc0, acc1;
#pragma unroll
  for (int i = 0; i < 8; ++i) { acc0[i] = 0.0f; acc1[i] = 0.0f; }
  gemm_bf16_tile(qu, DHEAD, kmat, DHEAD, DHEAD, m0w, n0w, lane, acc0, acc1);

  const int half = lane >> 4;
  const int colb = lane & 15;
#pragma unroll
  for (int r = 0; r < 8; ++r) {
    const int q = m0w + r + 8 * half;
#pragma unroll
    for (int f = 0; f < 2; ++f) {
      const int k = n0w + colb + 16 * f;
      const float acc = (f == 0) ? acc0[r] : acc1[r];
      float pos;
      if (k <= q)          pos = R[(size_t)q * SEQ + (k - q + SEQ - 1)];
      else if (k == q + 1) pos = 0.0f;
      else                 pos = R[(size_t)(q + 1) * SEQ + (k - q - 2)];
      Sc[(size_t)q * SEQ + k] = (acc + pos) * INV_SQRT_D;
    }
  }
}

// ---------------------------------------------------------------------------
// Row softmax over 1024 columns; writes bf16 attention for the A-side of
// the attn @ V WMMA GEMM.
// ---------------------------------------------------------------------------
__global__ __launch_bounds__(256) void softmax_row_kernel(
    const float* __restrict__ Sc, __bf16* __restrict__ attn) {
  __shared__ float red[256];
  const int row = blockIdx.x;
  const int tid = threadIdx.x;
  const float* sr = Sc + (size_t)row * SEQ;
  float v[4];
  float mx = -3.402823466e38f;
#pragma unroll
  for (int i = 0; i < 4; ++i) { v[i] = sr[tid + (i << 8)]; mx = fmaxf(mx, v[i]); }
  red[tid] = mx;
  __syncthreads();
  for (int off = 128; off > 0; off >>= 1) {
    if (tid < off) red[tid] = fmaxf(red[tid], red[tid + off]);
    __syncthreads();
  }
  mx = red[0];
  __syncthreads();
  float sum = 0.0f;
#pragma unroll
  for (int i = 0; i < 4; ++i) { v[i] = __expf(v[i] - mx); sum += v[i]; }
  red[tid] = sum;
  __syncthreads();
  for (int off = 128; off > 0; off >>= 1) {
    if (tid < off) red[tid] += red[tid + off];
    __syncthreads();
  }
  const float inv = 1.0f / red[0];
#pragma unroll
  for (int i = 0; i < 4; ++i)
    attn[(size_t)row * SEQ + tid + (i << 8)] = (__bf16)(v[i] * inv);
}

// ---------------------------------------------------------------------------
// ctx[q][dh] = sum_k attn[q][k] * v[k][dh]  for one (b,h).
// M = 1024, N = 64, K = 1024.  vT is [DH][S] so B-side loads are contiguous.
// Output bf16 into ctx (pre-offset by b,h; row stride DMODEL) ready to be the
// A-matrix of the final O projection.
// ---------------------------------------------------------------------------
__global__ __launch_bounds__(256) void wmma_av_kernel(
    const __bf16* __restrict__ attn, const __bf16* __restrict__ vT,
    __bf16* __restrict__ ctx) {
  const int lane = threadIdx.x & 31;
  const int wave = threadIdx.x >> 5;
  const int m0w = blockIdx.x * 64 + 16 * (wave & 3);
  const int n0w = 32 * (wave >> 2);  // N = 64 total
  v8f acc0, acc1;
#pragma unroll
  for (int i = 0; i < 8; ++i) { acc0[i] = 0.0f; acc1[i] = 0.0f; }
  gemm_bf16_tile(attn, SEQ, vT, SEQ, SEQ, m0w, n0w, lane, acc0, acc1);

  const int half = lane >> 4;
  const int colb = lane & 15;
#pragma unroll
  for (int r = 0; r < 8; ++r) {
    const int q = m0w + r + 8 * half;
    ctx[(size_t)q * DMODEL + n0w + colb]      = (__bf16)acc0[r];
    ctx[(size_t)q * DMODEL + n0w + colb + 16] = (__bf16)acc1[r];
  }
}

// ---------------------------------------------------------------------------
// Host-side orchestration
// ---------------------------------------------------------------------------
extern "C" void kernel_launch(void* const* d_in, const int* in_sizes, int n_in,
                              void* d_out, int out_size, void* d_ws, size_t ws_size,
                              hipStream_t stream) {
  (void)in_sizes; (void)n_in; (void)out_size; (void)ws_size;
  const float* inputs  = (const float*)d_in[0];
  const float* pos_emb = (const float*)d_in[1];
  const float* gamma   = (const float*)d_in[2];
  const float* beta    = (const float*)d_in[3];
  const float* Wq      = (const float*)d_in[4];
  const float* bq      = (const float*)d_in[5];
  const float* Wk      = (const float*)d_in[6];
  const float* bk      = (const float*)d_in[7];
  const float* Wv      = (const float*)d_in[8];
  const float* bv      = (const float*)d_in[9];
  const float* Wp      = (const float*)d_in[10];
  const float* u_bias  = (const float*)d_in[11];  // [H][DH] flat = 512
  const float* v_bias  = (const float*)d_in[12];
  const float* Wo      = (const float*)d_in[13];
  const float* bo      = (const float*)d_in[14];
  float* out = (float*)d_out;

  // --- workspace carve-up (~66 MB total) ---
  char* ws = (char*)d_ws;
  size_t off = 0;
  auto carve = [&](size_t bytes) -> void* {
    void* p = ws + off;
    off = (off + bytes + 255) & ~(size_t)255;
    return p;
  };
  const size_t MROWS = (size_t)BATCH * SEQ;                 // 8192
  __bf16* x_bf   = (__bf16*)carve(MROWS * DMODEL * 2);      // 8 MB
  __bf16* WTq    = (__bf16*)carve(DMODEL * DMODEL * 2);
  __bf16* WTk    = (__bf16*)carve(DMODEL * DMODEL * 2);
  __bf16* WTv    = (__bf16*)carve(DMODEL * DMODEL * 2);
  __bf16* WTp    = (__bf16*)carve(DMODEL * DMODEL * 2);
  __bf16* WTo    = (__bf16*)carve(DMODEL * DMODEL * 2);
  __bf16* pos_bf = (__bf16*)carve((size_t)SEQ * DMODEL * 2);
  __bf16* qu     = (__bf16*)carve(MROWS * DMODEL * 2);      // [B][H][S][DH]
  __bf16* qvb    = (__bf16*)carve(MROWS * DMODEL * 2);      // [B][H][S][DH]
  __bf16* kbuf   = (__bf16*)carve(MROWS * DMODEL * 2);      // [B][H][S][DH]
  __bf16* vT     = (__bf16*)carve(MROWS * DMODEL * 2);      // [B][H][DH][S]
  __bf16* p_bf   = (__bf16*)carve((size_t)NHEAD * SEQ * DHEAD * 2); // [H][S][DH]
  float*  Rbuf   = (float*)carve((size_t)SEQ * SEQ * 4);    // 4 MB, reused per (b,h)
  float*  Scbuf  = (float*)carve((size_t)SEQ * SEQ * 4);    // 4 MB, reused
  __bf16* attn   = (__bf16*)carve((size_t)SEQ * SEQ * 2);   // 2 MB, reused
  __bf16* ctx    = (__bf16*)carve(MROWS * DMODEL * 2);      // 8 MB

  // --- 1. weight conversion (transposed) + pos table conversion ---
  convert_transpose512_kernel<<<1024, 256, 0, stream>>>(Wq, WTq);
  convert_transpose512_kernel<<<1024, 256, 0, stream>>>(Wk, WTk);
  convert_transpose512_kernel<<<1024, 256, 0, stream>>>(Wv, WTv);
  convert_transpose512_kernel<<<1024, 256, 0, stream>>>(Wp, WTp);
  convert_transpose512_kernel<<<1024, 256, 0, stream>>>(Wo, WTo);
  convert_bf16_kernel<<<(SEQ * DMODEL + 255) / 256, 256, 0, stream>>>(
      pos_emb, pos_bf, SEQ * DMODEL);

  // --- 2. LayerNorm -> bf16 activations ---
  layernorm_bf16_kernel<<<MROWS, 256, 0, stream>>>(inputs, gamma, beta, x_bf);

  // --- 3. projections (WMMA bf16) ---
  dim3 gridP(MROWS / 64, DMODEL / 64);
  wmma_proj_gemm_kernel<<<gridP, 256, 0, stream>>>(
      x_bf, WTq, bq, u_bias, v_bias, qu, qvb, nullptr, 1);          // q+u, q+v
  wmma_proj_gemm_kernel<<<gridP, 256, 0, stream>>>(
      x_bf, WTk, bk, nullptr, nullptr, kbuf, nullptr, nullptr, 0);  // k
  wmma_proj_gemm_kernel<<<gridP, 256, 0, stream>>>(
      x_bf, WTv, bv, nullptr, nullptr, vT, nullptr, nullptr, 2);    // v (transposed)
  dim3 gridPos(SEQ / 64, DMODEL / 64);
  wmma_proj_gemm_kernel<<<gridPos, 256, 0, stream>>>(
      pos_bf, WTp, nullptr, nullptr, nullptr, p_bf, nullptr, nullptr, 0); // p

  // --- 4. attention per (b,h); 4 MB score buffers stay L2-resident ---
  dim3 gridS(SEQ / 64, SEQ / 64);
  for (int b = 0; b < BATCH; ++b) {
    for (int h = 0; h < NHEAD; ++h) {
      const size_t bh = (size_t)b * NHEAD + h;
      const __bf16* qu_bh = qu   + bh * SEQ * DHEAD;
      const __bf16* qv_bh = qvb  + bh * SEQ * DHEAD;
      const __bf16* k_bh  = kbuf + bh * SEQ * DHEAD;
      const __bf16* vT_bh = vT   + bh * DHEAD * SEQ;
      const __bf16* p_h   = p_bf + (size_t)h * SEQ * DHEAD;
      __bf16* ctx_bh = ctx + (size_t)b * SEQ * DMODEL + h * DHEAD;

      wmma_pos_raw_kernel<<<gridS, 256, 0, stream>>>(qv_bh, p_h, Rbuf);
      wmma_score_kernel<<<gridS, 256, 0, stream>>>(qu_bh, k_bh, Rbuf, Scbuf);
      softmax_row_kernel<<<SEQ, 256, 0, stream>>>(Scbuf, attn);
      wmma_av_kernel<<<dim3(SEQ / 64, 1), 256, 0, stream>>>(attn, vT_bh, ctx_bh);
    }
  }

  // --- 5. output projection -> f32 d_out ---
  wmma_proj_gemm_kernel<<<gridP, 256, 0, stream>>>(
      ctx, WTo, bo, nullptr, nullptr, nullptr, nullptr, out, 3);
}
